// CrossAttention_6287832121411
// MI455X (gfx1250) — matrix operand order repaired and checked by reading, not compiled, not run on hardware
//
#include <hip/hip_runtime.h>

// ---------------------------------------------------------------------------
// CrossAttention (sigmoid attention, attn is also an output) for MI455X/gfx1250
//
//   B=2, N=NY=2048, C=ENC=1024, H=16, HD=64, SCALE=1/8, bias=-log(2048)
//
// Roofline: 69 GFLOP total; mandatory f32 attn output = 537 MB -> ~23 us floor
// at 23.3 TB/s. All matmuls on v_wmma_f32_16x16x32_f16 (f32 accumulate).
// attn is written exactly once and consumed in-register (fused sigmoid + PV).
// The NY loop is split 2-way per query tile (8192 waves) for occupancy; the
// two partial PV results are summed inside the proj GEMM's A-fragment build.
// Workspace: Qh(8M f16)+Kh(8M)+VT(8M)+outp0(16M f32)+outp1(16M) = 56 MB.
// ---------------------------------------------------------------------------

typedef _Float16 v16h __attribute__((ext_vector_type(16)));
typedef _Float16 v8h  __attribute__((ext_vector_type(8)));
typedef float    v8f  __attribute__((ext_vector_type(8)));

#define B_   2
#define N_   2048
#define NY_  2048
#define C_   1024
#define H_   16
#define HD_  64
#define NSPLIT_ 2                     /* NY-splits per query tile in k_attn  */
#define ATTN_BIAS (-7.62461898616f)   /* -log(2048) */

__device__ __forceinline__ v8f wmma_f32_16x16x32_f16(v16h a, v16h b, v8f c) {
  return __builtin_amdgcn_wmma_f32_16x16x32_f16(false, a, false, b, (short)0, c,
                                                false, false);
}

// Build a 16x32 f16 A-fragment from a row-major f32 row (convert on load).
// Layout used here (not ISA table order): lane(0-15)=row M; elems 0-15 -> K = 16*g+e (contiguous)
__device__ __forceinline__ v16h a_frag_from_f32(const float* __restrict__ row,
                                                int k0, int g) {
  const float* p0 = row + k0 + 16 * g;
  const float* p1 = p0 + 8;
  v16h a;
#pragma unroll
  for (int i = 0; i < 8; ++i) {
    a[i]     = (_Float16)p0[i];
    a[i + 8] = (_Float16)p1[i];
  }
  return a;
}

// ---------------------------------------------------------------------------
// Stage 1a: Qh = f16(X @ Wq)   M=4096, N=1024, K=1024. One wave = 16x16 tile.
// ---------------------------------------------------------------------------
__global__ void __launch_bounds__(256) k_gemm_q(const float* __restrict__ X,
                                                const float* __restrict__ W,
                                                _Float16* __restrict__ Q) {
  const int lane = threadIdx.x & 31;
  const int wib  = threadIdx.x >> 5;
  const int w    = blockIdx.x * 8 + wib;
  const int nt   = w & 63;            // 1024/16 column tiles
  const int mt   = w >> 6;
  const int g    = lane >> 4;         // lane half
  const int r    = lane & 15;
  const float* xrow = X + (size_t)(mt * 16 + r) * C_;
  const int col = nt * 16 + r;
  v8f acc = {};
  for (int kc = 0; kc < C_; kc += 32) {
    v16h a = a_frag_from_f32(xrow, kc, g);
    // B-fragment: lane=col, K = kc + 16*g + e (lane-coalesced over W rows)
    const float* wp = W + (size_t)(kc + 16 * g) * C_ + col;
    v16h bf;
#pragma unroll
    for (int i = 0; i < 16; ++i) bf[i] = (_Float16)wp[(size_t)i * C_];
    acc = wmma_f32_16x16x32_f16(a, bf, acc);
  }
#pragma unroll
  for (int j = 0; j < 8; ++j)
    Q[(size_t)(mt * 16 + j + 8 * g) * C_ + col] = (_Float16)acc[j];
}

// ---------------------------------------------------------------------------
// Stage 1b: kv = Y @ Wkv  M=4096, N=2048, K=1024.
//   cols [0,1024)   -> Kh f16, natural [b*NY+ny, C] layout
//   cols [1024,2048)-> V transposed via LDS to VT[b][h][hd][ny] f16
// ---------------------------------------------------------------------------
__global__ void __launch_bounds__(256) k_gemm_kv(const float* __restrict__ Y,
                                                 const float* __restrict__ W,
                                                 _Float16* __restrict__ K,
                                                 _Float16* __restrict__ VT) {
  __shared__ _Float16 t[8][256];      // per-wave 16x16 transpose tile
  const int lane = threadIdx.x & 31;
  const int wib  = threadIdx.x >> 5;
  const int w    = blockIdx.x * 8 + wib;
  const int nt   = w & 127;           // 2048/16 column tiles
  const int mt   = w >> 7;
  const int g    = lane >> 4;
  const int r    = lane & 15;
  const float* yrow = Y + (size_t)(mt * 16 + r) * C_;
  const int col = nt * 16 + r;
  v8f acc = {};
  for (int kc = 0; kc < C_; kc += 32) {
    v16h a = a_frag_from_f32(yrow, kc, g);
    const float* wp = W + (size_t)(kc + 16 * g) * (2 * C_) + col;
    v16h bf;
#pragma unroll
    for (int i = 0; i < 16; ++i) bf[i] = (_Float16)wp[(size_t)i * (2 * C_)];
    acc = wmma_f32_16x16x32_f16(a, bf, acc);
  }
  if (nt < 64) {                      // wave-uniform branch (after all WMMAs)
#pragma unroll
    for (int j = 0; j < 8; ++j)
      K[(size_t)(mt * 16 + j + 8 * g) * C_ + col] = (_Float16)acc[j];
  } else {
    // store transposed into LDS: t[hd_local=r][ny_local=j+8*g]
#pragma unroll
    for (int j = 0; j < 8; ++j)
      t[wib][r * 16 + j + 8 * g] = (_Float16)acc[j];
    v8h pack = *(const v8h*)&t[wib][r * 16 + 8 * g];  // hd=r, ny = 8*g..8*g+7
    const int c    = col - C_;        // 0..1023
    const int h    = c >> 6;
    const int hd   = c & 63;
    const int row0 = mt * 16;
    const int b    = row0 >> 11;
    const int ny   = (row0 & (NY_ - 1)) + 8 * g;
    *(v8h*)(VT + (size_t)((b * H_ + h) * HD_ + hd) * NY_ + ny) = pack;
  }
}

// ---------------------------------------------------------------------------
// Stage 2: fused sigmoid attention. One wave = (b,h,16-row q tile, NY half).
// Per 32 NY columns: 4 WMMA (QK^T, K=64) -> sigmoid -> f32 attn store (once)
// -> LDS re-stage of probs to A-fragment layout -> 4 WMMA (P@V) accumulate.
// Partial PV sums go to outp[s]; summed later inside the proj GEMM.
// ---------------------------------------------------------------------------
__global__ void __launch_bounds__(256) k_attn(const _Float16* __restrict__ Q,
                                              const _Float16* __restrict__ K,
                                              const _Float16* __restrict__ VT,
                                              float* __restrict__ attn,
                                              float* __restrict__ outp) {
  __shared__ _Float16 pt[8][512];     // per-wave 16x32 f16 P tile (1 KB each)
  const int lane = threadIdx.x & 31;
  const int wib  = threadIdx.x >> 5;
  const int w    = blockIdx.x * 8 + wib;   // 0..8191
  const int s    = w & (NSPLIT_ - 1);      // NY half
  const int tile = w >> 1;                 // 0..4095
  const int mt   = tile & 127;             // N/16
  const int bh   = tile >> 7;
  const int h    = bh & 15;
  const int b    = bh >> 4;
  const int g    = lane >> 4;
  const int r    = lane & 15;
  const int ny_beg = s * (NY_ / NSPLIT_);
  const int ny_end = ny_beg + NY_ / NSPLIT_;

  const _Float16* qrow = Q + (size_t)(b * N_ + mt * 16 + r) * C_ + h * HD_;
  const _Float16* kb   = K + (size_t)(b * NY_) * C_ + h * HD_;
  const _Float16* vb   = VT + (size_t)((b * H_ + h) * HD_) * NY_;
  float* ab = attn + ((size_t)(b * H_ + h) * N_ + mt * 16) * (size_t)NY_;

  // q A-fragments for hd chunks [0,32) and [32,64): contiguous 16B per lane
  v16h qa0, qa1;
  {
    v8h l0 = *(const v8h*)(qrow + 16 * g);
    v8h h0 = *(const v8h*)(qrow + 8 + 16 * g);
    v8h l1 = *(const v8h*)(qrow + 32 + 16 * g);
    v8h h1 = *(const v8h*)(qrow + 40 + 16 * g);
#pragma unroll
    for (int i = 0; i < 8; ++i) {
      qa0[i] = l0[i]; qa0[i + 8] = h0[i];
      qa1[i] = l1[i]; qa1[i + 8] = h1[i];
    }
  }

  v8f acc0 = {}, acc1 = {}, acc2 = {}, acc3 = {};
  _Float16* pl = &pt[wib][0];

  for (int ny0 = ny_beg; ny0 < ny_end; ny0 += 32) {
    // prefetch next iteration's K rows / V chunk (uniform branch; L2 hints)
    if (ny0 + 32 < ny_end) {
      __builtin_prefetch(kb + (size_t)(ny0 + 32 + r) * C_, 0, 0);
      __builtin_prefetch(vb + (size_t)r * NY_ + ny0 + 32 + 16 * g, 0, 0);
    }
#pragma unroll
    for (int ct = 0; ct < 2; ++ct) {
      // K^T B-fragments: lane=ny column, K=hd contiguous (32B per lane)
      const _Float16* krow = kb + (size_t)(ny0 + ct * 16 + r) * C_;
      v16h b0 = *(const v16h*)(krow + 16 * g);
      v16h b1 = *(const v16h*)(krow + 32 + 16 * g);
      v8f sc = {};
      sc = wmma_f32_16x16x32_f16(qa0, b0, sc);
      sc = wmma_f32_16x16x32_f16(qa1, b1, sc);
#pragma unroll
      for (int j = 0; j < 8; ++j) {
        float z = sc[j] * 0.125f + ATTN_BIAS;
        float p = 1.0f / (1.0f + __expf(-z));
        const int m = j + 8 * g;
        ab[(size_t)m * NY_ + ny0 + ct * 16 + r] = p;   // attn output (coalesced)
        pl[m * 32 + ct * 16 + r] = (_Float16)p;        // stage for PV A-frag
      }
    }
    // re-read probs in A-fragment layout (same-wave DS ops are in-order)
    v16h pa;
    {
      v8h lo = *(const v8h*)(pl + r * 32 + 16 * g);
      v8h hi = *(const v8h*)(pl + r * 32 + 8 + 16 * g);
#pragma unroll
      for (int i = 0; i < 8; ++i) { pa[i] = lo[i]; pa[i + 8] = hi[i]; }
    }
    // V B-fragments from VT: lane=hd column, K=ny contiguous (32B per lane)
    const _Float16* v0 = vb + (size_t)r * NY_ + ny0 + 16 * g;
    acc0 = wmma_f32_16x16x32_f16(pa, *(const v16h*)(v0),                  acc0);
    acc1 = wmma_f32_16x16x32_f16(pa, *(const v16h*)(v0 + (size_t)16*NY_), acc1);
    acc2 = wmma_f32_16x16x32_f16(pa, *(const v16h*)(v0 + (size_t)32*NY_), acc2);
    acc3 = wmma_f32_16x16x32_f16(pa, *(const v16h*)(v0 + (size_t)48*NY_), acc3);
  }

  float* ob = outp + (size_t)s * B_ * N_ * C_ +
              (size_t)(b * N_ + mt * 16) * C_ + h * HD_;
#pragma unroll
  for (int j = 0; j < 8; ++j) {
    const int m = j + 8 * g;
    ob[(size_t)m * C_ + r]      = acc0[j];
    ob[(size_t)m * C_ + 16 + r] = acc1[j];
    ob[(size_t)m * C_ + 32 + r] = acc2[j];
    ob[(size_t)m * C_ + 48 + r] = acc3[j];
  }
}

// ---------------------------------------------------------------------------
// Stage 3: out = (outp0 + outp1) @ Wproj + bproj  (sum partials in A-frag)
// ---------------------------------------------------------------------------
__global__ void __launch_bounds__(256) k_gemm_proj(const float* __restrict__ X0,
                                                   const float* __restrict__ X1,
                                                   const float* __restrict__ W,
                                                   const float* __restrict__ bias,
                                                   float* __restrict__ O) {
  const int lane = threadIdx.x & 31;
  const int wib  = threadIdx.x >> 5;
  const int w    = blockIdx.x * 8 + wib;
  const int nt   = w & 63;
  const int mt   = w >> 6;
  const int g    = lane >> 4;
  const int r    = lane & 15;
  const size_t rowoff = (size_t)(mt * 16 + r) * C_;
  const float* xrow0 = X0 + rowoff;
  const float* xrow1 = X1 + rowoff;
  const int col = nt * 16 + r;
  v8f acc = {};
  for (int kc = 0; kc < C_; kc += 32) {
    // A-fragment = f16(partial0 + partial1)
    const float* p0a = xrow0 + kc + 16 * g;
    const float* p0b = p0a + 8;
    const float* p1a = xrow1 + kc + 16 * g;
    const float* p1b = p1a + 8;
    v16h a;
#pragma unroll
    for (int i = 0; i < 8; ++i) {
      a[i]     = (_Float16)(p0a[i] + p1a[i]);
      a[i + 8] = (_Float16)(p0b[i] + p1b[i]);
    }
    const float* wp = W + (size_t)(kc + 16 * g) * C_ + col;
    v16h bf;
#pragma unroll
    for (int i = 0; i < 16; ++i) bf[i] = (_Float16)wp[(size_t)i * C_];
    acc = wmma_f32_16x16x32_f16(a, bf, acc);
  }
  const float bv = bias[col];
#pragma unroll
  for (int j = 0; j < 8; ++j)
    O[(size_t)(mt * 16 + j + 8 * g) * C_ + col] = acc[j] + bv;
}

// ---------------------------------------------------------------------------
extern "C" void kernel_launch(void* const* d_in, const int* in_sizes, int n_in,
                              void* d_out, int out_size, void* d_ws, size_t ws_size,
                              hipStream_t stream) {
  (void)in_sizes; (void)n_in; (void)out_size; (void)ws_size;
  const float* x   = (const float*)d_in[0];   // [B,N,C]
  const float* y   = (const float*)d_in[1];   // [B,NY,ENC]
  const float* Wq  = (const float*)d_in[2];   // [C,C]
  const float* Wkv = (const float*)d_in[3];   // [ENC,2C]
  const float* Wp  = (const float*)d_in[4];   // [C,C]
  const float* bp  = (const float*)d_in[5];   // [C]

  float* out  = (float*)d_out;                       // [B,N,C]
  float* attn = out + (size_t)B_ * N_ * C_;          // [B,H,N,NY]

  char* ws = (char*)d_ws;                            // 56 MB total
  _Float16* Qh = (_Float16*)ws; ws += (size_t)B_ * N_  * C_ * sizeof(_Float16);
  _Float16* Kh = (_Float16*)ws; ws += (size_t)B_ * NY_ * C_ * sizeof(_Float16);
  _Float16* VT = (_Float16*)ws; ws += (size_t)B_ * NY_ * C_ * sizeof(_Float16);
  float* outp  = (float*)ws;    // NSPLIT_ x [B,N,C] f32 partial PV sums
  float* outp1 = outp + (size_t)B_ * N_ * C_;

  // (4096x1024)x(1024)  -> 16384 waves = 2048 blocks of 8 waves
  k_gemm_q   <<<2048, 256, 0, stream>>>(x, Wq, Qh);
  // (4096x2048)x(1024)  -> 32768 waves
  k_gemm_kv  <<<4096, 256, 0, stream>>>(y, Wkv, Kh, VT);
  // 2*16*128 q-tiles x 2 NY-halves = 8192 waves
  k_attn     <<<1024, 256, 0, stream>>>(Qh, Kh, VT, attn, outp);
  // (4096x1024)x(1024), sums the two PV partials on the fly
  k_gemm_proj<<<2048, 256, 0, stream>>>(outp, outp1, Wp, bp, out);
}
